// Attention_4552665334102
// MI455X (gfx1250) — matrix operand order repaired and checked by reading, not compiled, not run on hardware
//
#include <hip/hip_runtime.h>

typedef __attribute__((ext_vector_type(16))) __bf16 v16bf;
typedef __attribute__((ext_vector_type(8)))  float  v8f;

#define B_   4
#define N_   2048
#define E_   1024   // QUERY_DIM == INNER
#define H_   16
#define D_   64

// ---------------------------------------------------------------------------
// CDNA5 async global->LDS copy (ASYNCcnt) with safe fallback
// ---------------------------------------------------------------------------
#if defined(__HIP_DEVICE_COMPILE__) && defined(__has_builtin)
#if __has_builtin(__builtin_amdgcn_global_load_async_to_lds_b128) && \
    __has_builtin(__builtin_amdgcn_s_wait_asynccnt)
#define HAVE_ASYNC 1
#endif
#endif
#ifndef HAVE_ASYNC
#define HAVE_ASYNC 0
#endif

typedef int v4i_vs __attribute__((__vector_size__(4 * sizeof(int))));

__device__ __forceinline__ void cp16_async(void* lds, const void* g) {
#if HAVE_ASYNC
    __builtin_amdgcn_global_load_async_to_lds_b128(
        (__attribute__((address_space(1))) v4i_vs*)(unsigned long long)g,
        (__attribute__((address_space(3))) v4i_vs*)(unsigned)(unsigned long long)lds,
        0, 0);
#else
    *(uint4*)lds = *(const uint4*)g;
#endif
}
__device__ __forceinline__ void async_join() {
#if HAVE_ASYNC
    __builtin_amdgcn_s_wait_asynccnt(0);
#endif
}

// A-matrix (16x32 bf16) K index for vector element (i = elem/2, e = elem&1)
__device__ __forceinline__ int a_kmap(int i, int e, int khalf) {
    int base = (i < 4) ? (2 * i) : (16 + 2 * (i - 4));
    return base + 8 * khalf + e;
}
// B-matrix (32x16 bf16) K index
__device__ __forceinline__ int b_kmap(int i, int e, int khalf) {
    return ((i < 4) ? (2 * i) : (16 + 2 * (i - 4))) + 8 * khalf + e;
}

__device__ __forceinline__ v8f wmma_bf16(v16bf a, v16bf b, v8f c) {
    return __builtin_amdgcn_wmma_f32_16x16x32_bf16(false, a, false, b,
                                                   (short)0, c, false, false);
}

// ---------------------------------------------------------------------------
// C[128x64 tile] = A(f32, MxK row-major) * B(f32, KxN row-major) -> bf16
// transpose==0: split-head [b][h][n][64] layout (Q, K; scale folds 1/sqrt(d))
// transpose==1: head-transposed [b][h][64][n] layout (V), 16B packed stores
// 4 waves; each wave computes a 64x32 sub-tile (8 WMMAs / K-step)
// ---------------------------------------------------------------------------
__global__ __launch_bounds__(128)
void proj_gemm_kernel(const float* __restrict__ Xa, const float* __restrict__ W,
                      __bf16* __restrict__ dst, float scale, int transpose)
{
    __shared__ __align__(16) __bf16 As[128 * 32];  // [m][k]
    __shared__ __align__(16) __bf16 Bs[64 * 32];   // transposed: [n][k]

    const int tid  = threadIdx.x;
    const int lane = tid & 31;
    const int wave = tid >> 5;
    const int m0 = blockIdx.x * 128;
    const int n0 = blockIdx.y * 64;

    const int l15   = lane & 15;
    const int khalf = lane >> 4;
    const int mbase = (wave & 1) * 64;
    const int nbase = (wave >> 1) * 32;

    v8f acc[4][2] = {};

    const int b_k    = tid & 31;
    const int b_nseg = (tid >> 5) * 16;

    for (int k0 = 0; k0 < E_; k0 += 32) {
        if (k0 + 32 < E_) {
            __builtin_prefetch(Xa + (size_t)(m0 + tid) * E_ + k0 + 32, 0, 1);
            __builtin_prefetch(W + (size_t)(k0 + 32 + b_k) * E_ + n0 + b_nseg, 0, 1);
        }
        __syncthreads();
        { // A tile (f32 -> bf16): one row per thread
            const float4* src = (const float4*)(Xa + (size_t)(m0 + tid) * E_ + k0);
            #pragma unroll
            for (int q = 0; q < 8; ++q) {
                float4 v = src[q];
                int o = tid * 32 + q * 4;
                As[o + 0] = (__bf16)v.x; As[o + 1] = (__bf16)v.y;
                As[o + 2] = (__bf16)v.z; As[o + 3] = (__bf16)v.w;
            }
        }
        { // B tile, stored transposed so (k,k+1) pairs are contiguous
            const float4* src = (const float4*)(W + (size_t)(k0 + b_k) * E_ + n0 + b_nseg);
            #pragma unroll
            for (int q = 0; q < 4; ++q) {
                float4 v = src[q];
                int nb = b_nseg + q * 4;
                Bs[(nb + 0) * 32 + b_k] = (__bf16)v.x;
                Bs[(nb + 1) * 32 + b_k] = (__bf16)v.y;
                Bs[(nb + 2) * 32 + b_k] = (__bf16)v.z;
                Bs[(nb + 3) * 32 + b_k] = (__bf16)v.w;
            }
        }
        __syncthreads();

        v16bf bf[2];
        #pragma unroll
        for (int ni = 0; ni < 2; ++ni)
            #pragma unroll
            for (int j = 0; j < 16; ++j)
                bf[ni][j] = Bs[(nbase + ni * 16 + l15) * 32 + b_kmap(j >> 1, j & 1, khalf)];
        #pragma unroll
        for (int mi = 0; mi < 4; ++mi) {
            v16bf af;
            #pragma unroll
            for (int j = 0; j < 16; ++j)
                af[j] = As[(mbase + mi * 16 + l15) * 32 + a_kmap(j >> 1, j & 1, khalf)];
            #pragma unroll
            for (int ni = 0; ni < 2; ++ni)
                acc[mi][ni] = wmma_bf16(af, bf[ni], acc[mi][ni]);
        }
    }

    if (!transpose) {
        #pragma unroll
        for (int mi = 0; mi < 4; ++mi)
            #pragma unroll
            for (int ni = 0; ni < 2; ++ni) {
                const int gn = n0 + nbase + ni * 16 + l15;
                const int hh = gn >> 6, dd = gn & 63;
                #pragma unroll
                for (int r = 0; r < 8; ++r) {
                    int gm = m0 + mbase + mi * 16 + r + 8 * khalf;
                    int bidx = gm >> 11, nn = gm & 2047;
                    size_t o = ((size_t)(bidx * H_ + hh) * N_ + nn) * (size_t)D_ + dd;
                    dst[o] = (__bf16)(acc[mi][ni][r] * scale);
                }
            }
    } else {
        // [b][h][d][n] layout: the 8 C rows per lane are consecutive n -> 16B store
        #pragma unroll
        for (int mi = 0; mi < 4; ++mi)
            #pragma unroll
            for (int ni = 0; ni < 2; ++ni) {
                const int gn = n0 + nbase + ni * 16 + l15;
                const int hh = gn >> 6, dd = gn & 63;
                const int gmb = m0 + mbase + mi * 16 + 8 * khalf;  // r = 0
                const int bidx = gmb >> 11, nn = gmb & 2047;
                union { __bf16 e[8]; uint4 u; } pk;
                #pragma unroll
                for (int r = 0; r < 8; ++r)
                    pk.e[r] = (__bf16)(acc[mi][ni][r] * scale);
                size_t o = ((size_t)(bidx * H_ + hh) * D_ + dd) * (size_t)N_ + nn;
                *(uint4*)(dst + o) = pk.u;
            }
    }
}

// ---------------------------------------------------------------------------
// Flash attention: 1 WG (4 waves) per 64 query rows of one (b,h).
// 64-key tiles streamed via async global->LDS; online softmax in registers.
// Q,K: [b][h][n][64] bf16; V: [b][h][64][n] bf16 (pre-transposed).
// ---------------------------------------------------------------------------
__global__ __launch_bounds__(128)
void attn_kernel(const __bf16* __restrict__ Qb, const __bf16* __restrict__ Kb,
                 const __bf16* __restrict__ Vtg, __bf16* __restrict__ Ob)
{
    __shared__ __align__(16) __bf16 Ks[64 * 64];       // [key][d]
    __shared__ __align__(16) __bf16 Vt[64 * 64];       // [d][key]
    __shared__ __align__(16) __bf16 Ps[4][16 * 64];    // per-wave P staging [m][j]

    const int tid  = threadIdx.x;
    const int lane = tid & 31;
    const int wave = tid >> 5;
    const int q0 = blockIdx.x * 64;
    const int bh = blockIdx.y;
    const size_t seqbase = (size_t)bh * N_ * D_;

    const int l15   = lane & 15;
    const int khalf = lane >> 4;

    // preload Q A-fragments (1/sqrt(d) folded in at projection time)
    v16bf qf[2];
    {
        const __bf16* qrow = Qb + seqbase + (size_t)(q0 + wave * 16 + l15) * D_;
        #pragma unroll
        for (int c = 0; c < 2; ++c)
            #pragma unroll
            for (int j = 0; j < 16; ++j)
                qf[c][j] = qrow[c * 32 + a_kmap(j >> 1, j & 1, khalf)];
    }

    float mrun[8], lrun[8];
    #pragma unroll
    for (int r = 0; r < 8; ++r) { mrun[r] = -3.0e38f; lrun[r] = 0.0f; }
    v8f oacc[4] = {};

    for (int j0 = 0; j0 < N_; j0 += 64) {
        const __bf16* ksrc = Kb  + seqbase + (size_t)j0 * D_;   // contiguous 8KB
        const __bf16* vsrc = Vtg + seqbase + j0;                // 64 rows x 128B
        if (j0 + 64 < N_) __builtin_prefetch(ksrc + 64 * D_ + tid * 32, 0, 1);
        __syncthreads();
        #pragma unroll
        for (int i = 0; i < 4; ++i) {
            int u = tid + i * 128;                               // 0..511 (16B units)
            cp16_async((char*)Ks + u * 16, (const char*)ksrc + u * 16);
            int d = u >> 3, off = (u & 7) * 8;                   // V row d, elem off
            cp16_async(Vt + d * 64 + off, vsrc + (size_t)d * N_ + off);
        }
        async_join();
        __syncthreads();

        // S = Q * K^T : 4 key-subtiles of 16, contraction d=64 split into 2x32
        v8f sacc[4];
        #pragma unroll
        for (int jt = 0; jt < 4; ++jt) {
            v8f sc = {};
            #pragma unroll
            for (int c = 0; c < 2; ++c) {
                v16bf kf;
                #pragma unroll
                for (int j = 0; j < 16; ++j)
                    kf[j] = Ks[(jt * 16 + l15) * 64 + c * 32 + b_kmap(j >> 1, j & 1, khalf)];
                sc = wmma_bf16(qf[c], kf, sc);
            }
            sacc[jt] = sc;
        }

        // online softmax per row (row = r + 8*khalf; uniform across 16-lane group)
        #pragma unroll
        for (int r = 0; r < 8; ++r) {
            float t = fmaxf(fmaxf(sacc[0][r], sacc[1][r]),
                            fmaxf(sacc[2][r], sacc[3][r]));
            t = fmaxf(t, __shfl_xor(t, 1));
            t = fmaxf(t, __shfl_xor(t, 2));
            t = fmaxf(t, __shfl_xor(t, 4));
            t = fmaxf(t, __shfl_xor(t, 8));
            float mn    = fmaxf(mrun[r], t);
            float alpha = __expf(mrun[r] - mn);
            float p[4], rs = 0.0f;
            #pragma unroll
            for (int jt = 0; jt < 4; ++jt) {
                p[jt] = __expf(sacc[jt][r] - mn);
                rs += p[jt];
            }
            rs += __shfl_xor(rs, 1);
            rs += __shfl_xor(rs, 2);
            rs += __shfl_xor(rs, 4);
            rs += __shfl_xor(rs, 8);
            lrun[r] = lrun[r] * alpha + rs;
            mrun[r] = mn;
            #pragma unroll
            for (int t4 = 0; t4 < 4; ++t4) oacc[t4][r] *= alpha;
            int prow = (r + 8 * khalf) * 64;
            #pragma unroll
            for (int jt = 0; jt < 4; ++jt)
                Ps[wave][prow + jt * 16 + l15] = (__bf16)p[jt];
        }

        // re-read P in A-layout (same-wave DS ops are in-order)
        v16bf pf[2];
        #pragma unroll
        for (int c = 0; c < 2; ++c)
            #pragma unroll
            for (int j = 0; j < 16; ++j)
                pf[c][j] = Ps[wave][l15 * 64 + c * 32 + a_kmap(j >> 1, j & 1, khalf)];

        // O += P * V  (4 d-tiles of 16, contraction j=64 split into 2x32)
        #pragma unroll
        for (int t4 = 0; t4 < 4; ++t4)
            #pragma unroll
            for (int c = 0; c < 2; ++c) {
                v16bf vf;
                #pragma unroll
                for (int j = 0; j < 16; ++j)
                    vf[j] = Vt[(t4 * 16 + l15) * 64 + c * 32 + b_kmap(j >> 1, j & 1, khalf)];
                oacc[t4] = wmma_bf16(pf[c], vf, oacc[t4]);
            }
    }

    // normalize and write O as [b][n][h*64+d] bf16 (row-major for final GEMM)
    const int bb = bh >> 4, hh = bh & 15;
    #pragma unroll
    for (int t4 = 0; t4 < 4; ++t4)
        #pragma unroll
        for (int r = 0; r < 8; ++r) {
            int row = q0 + wave * 16 + r + 8 * khalf;
            int col = hh * 64 + t4 * 16 + l15;
            Ob[((size_t)bb * N_ + row) * E_ + col] = (__bf16)(oacc[t4][r] / lrun[r]);
        }
}

// ---------------------------------------------------------------------------
// out[128x64 tile] = O(bf16) * Wo(f32) + bo, f32 output; async A-tile copies
// ---------------------------------------------------------------------------
__global__ __launch_bounds__(128)
void out_gemm_kernel(const __bf16* __restrict__ Ao, const float* __restrict__ W,
                     const float* __restrict__ bias, float* __restrict__ out)
{
    __shared__ __align__(16) __bf16 As[128 * 32];
    __shared__ __align__(16) __bf16 Bs[64 * 32];

    const int tid  = threadIdx.x;
    const int lane = tid & 31;
    const int wave = tid >> 5;
    const int m0 = blockIdx.x * 128;
    const int n0 = blockIdx.y * 64;

    const int l15   = lane & 15;
    const int khalf = lane >> 4;
    const int mbase = (wave & 1) * 64;
    const int nbase = (wave >> 1) * 32;

    v8f acc[4][2] = {};

    const int b_k    = tid & 31;
    const int b_nseg = (tid >> 5) * 16;

    for (int k0 = 0; k0 < E_; k0 += 32) {
        if (k0 + 32 < E_)
            __builtin_prefetch(W + (size_t)(k0 + 32 + b_k) * E_ + n0 + b_nseg, 0, 1);
        __syncthreads();
        { // A tile: bf16 straight async copies (128 rows x 64B)
            #pragma unroll
            for (int i = 0; i < 4; ++i) {
                int u = tid + i * 128;                  // 0..511 (16B units)
                int row = u >> 2, koff = (u & 3) * 8;   // 4 units per 64B row
                cp16_async(As + row * 32 + koff,
                           Ao + (size_t)(m0 + row) * E_ + k0 + koff);
            }
        }
        { // B tile transposed (f32 -> bf16)
            const float4* src = (const float4*)(W + (size_t)(k0 + b_k) * E_ + n0 + b_nseg);
            #pragma unroll
            for (int q = 0; q < 4; ++q) {
                float4 v = src[q];
                int nb = b_nseg + q * 4;
                Bs[(nb + 0) * 32 + b_k] = (__bf16)v.x;
                Bs[(nb + 1) * 32 + b_k] = (__bf16)v.y;
                Bs[(nb + 2) * 32 + b_k] = (__bf16)v.z;
                Bs[(nb + 3) * 32 + b_k] = (__bf16)v.w;
            }
        }
        async_join();
        __syncthreads();

        v16bf bf[2];
        #pragma unroll
        for (int ni = 0; ni < 2; ++ni)
            #pragma unroll
            for (int j = 0; j < 16; ++j)
                bf[ni][j] = Bs[(nbase + ni * 16 + l15) * 32 + b_kmap(j >> 1, j & 1, khalf)];
        #pragma unroll
        for (int mi = 0; mi < 4; ++mi) {
            v16bf af;
            #pragma unroll
            for (int j = 0; j < 16; ++j)
                af[j] = As[(mbase + mi * 16 + l15) * 32 + a_kmap(j >> 1, j & 1, khalf)];
            #pragma unroll
            for (int ni = 0; ni < 2; ++ni)
                acc[mi][ni] = wmma_bf16(af, bf[ni], acc[mi][ni]);
        }
    }

    #pragma unroll
    for (int ni = 0; ni < 2; ++ni) {
        const int gn = n0 + nbase + ni * 16 + l15;
        const float bv = bias[gn];
        #pragma unroll
        for (int mi = 0; mi < 4; ++mi)
            #pragma unroll
            for (int r = 0; r < 8; ++r) {
                int gm = m0 + mbase + mi * 16 + r + 8 * khalf;
                out[(size_t)gm * E_ + gn] = acc[mi][ni][r] + bv;
            }
    }
}

// ---------------------------------------------------------------------------
extern "C" void kernel_launch(void* const* d_in, const int* in_sizes, int n_in,
                              void* d_out, int out_size, void* d_ws, size_t ws_size,
                              hipStream_t stream)
{
    (void)in_sizes; (void)n_in; (void)out_size; (void)ws_size;
    const float* x  = (const float*)d_in[0];
    const float* Wq = (const float*)d_in[1];
    const float* Wk = (const float*)d_in[2];
    const float* Wv = (const float*)d_in[3];
    const float* Wo = (const float*)d_in[4];
    const float* bo = (const float*)d_in[5];
    float* out = (float*)d_out;

    __bf16* ws = (__bf16*)d_ws;
    const size_t per = (size_t)B_ * H_ * N_ * D_;   // 8M elements
    __bf16* Qb = ws;
    __bf16* Kb = Qb + per;
    __bf16* Vb = Kb + per;                           // [b][h][64][n] (transposed)
    __bf16* Ob = Vb + per;                           // [B*N_][E_]

    dim3 blk(128, 1, 1);
    dim3 gproj((B_ * N_) / 128, E_ / 64, 1);         // 64 x 16
    proj_gemm_kernel<<<gproj, blk, 0, stream>>>(x, Wq, Qb, 0.125f, 0); // 1/sqrt(64)
    proj_gemm_kernel<<<gproj, blk, 0, stream>>>(x, Wk, Kb, 1.0f, 0);
    proj_gemm_kernel<<<gproj, blk, 0, stream>>>(x, Wv, Vb, 1.0f, 1);

    dim3 gattn(N_ / 64, B_ * H_, 1);                 // 32 x 64
    attn_kernel<<<gattn, blk, 0, stream>>>(Qb, Kb, Vb, Ob);

    dim3 gout((B_ * N_) / 128, E_ / 64, 1);          // 64 x 16
    out_gemm_kernel<<<gout, blk, 0, stream>>>(Ob, Wo, bo, out);
}
